// AttentionModule_3255585210805
// MI455X (gfx1250) — matrix operand order repaired and checked by reading, not compiled, not run on hardware
//
#include <hip/hip_runtime.h>
#include <hip/hip_bf16.h>
#include <math.h>

// ---------------------------------------------------------------------------
// Types for CDNA5 WMMA (wave32): D(16x16 f32) = A(16x32 bf16) * B(32x16 bf16) + C
// ---------------------------------------------------------------------------
typedef __bf16 bf16_t;
typedef __attribute__((ext_vector_type(4)))  __bf16 v4bf;
typedef __attribute__((ext_vector_type(8)))  __bf16 v8bf;
typedef __attribute__((ext_vector_type(16))) __bf16 v16bf;
typedef __attribute__((ext_vector_type(8)))  float  v8f;
typedef __attribute__((ext_vector_type(4)))  float  v4f;

// TDM descriptor operand types (6-arg clang-23 form:
//  (uint32x4 g0, int32x8 g1, int32x4, int32x4, int32x8, i32 cpol))
typedef __attribute__((ext_vector_type(4))) unsigned int v4u_t;
typedef __attribute__((ext_vector_type(8))) int          v8i_t;
typedef __attribute__((ext_vector_type(4))) int          v4i_t;

#if defined(__has_builtin)
#  if __has_builtin(__builtin_amdgcn_tensor_load_to_lds) && \
      __has_builtin(__builtin_amdgcn_s_wait_tensorcnt)
#    define USE_TDM 1
#  endif
#endif
#ifndef USE_TDM
#  define USE_TDM 0
#endif

// Problem constants (B=2, N=M=F=1024, E=64, G=16, dg=64, DIM_Q=DIM_OUT=1024)
#define BB   2
#define NN   1024
#define MM   1024
#define FF   1024
#define EE   64
#define GG   16

// Workspace layout (bytes, all 256-aligned). Total ~211 MB.
#define ROIV_OFF      0LL            // roi_feat bf16           4 MiB
#define QW_OFF        4194304LL      // q_w bf16                2 MiB
#define KW_OFF        6291456LL      // k_w bf16                2 MiB
#define CW_OFF        8388608LL      // conv_w bf16             2 MiB
#define POSW_OFF      10485760LL     // pos_w bf16              2 KiB
#define QB16_OFF      10487808LL     // q bf16 (B,N,1024)       4 MiB
#define KB16_OFF      14682112LL     // k bf16 (B,M,1024)       4 MiB
#define WGT_OFF       18876416LL     // weighted f32 (B,N,G,M)  128 MiB ; reused for p (bf16, stride-2048)
#define OUTT_OFF      153094144LL    // out_t bf16 (B,N,G,F)    64 MiB

// ---------------------------------------------------------------------------
// Fragment loaders following the CDNA5 ISA VGPR layouts (CDNA5 ISA):
//  A (16x32 bf16): lane<16 -> M=lane, elems {K0..7, K16..23}; lane>=16 -> M=lane-16, {K8..15, K24..31}
//  B (32x16 bf16): lane<16 -> N=lane, elems {K0..7, K16..23};          lane>=16 -> N=lane-16, {K8..15, K24..31}
//  C/D (16x16 f32): VGPR e: lane<16 -> M=e, N=lane; lane>=16 -> M=8+e, N=lane-16
// Source tiles are row-major with K contiguous (A: [M][K], B: [N][K]).
// ---------------------------------------------------------------------------
__device__ __forceinline__ v16bf load_a_frag(const bf16_t* base, int rowStride,
                                             int kbase, int lane) {
  const int hi = lane >> 4;
  const bf16_t* p = base + (size_t)(lane & 15) * rowStride + kbase + hi * 8;
  v8bf lo = *(const v8bf*)p;          // K = kbase + hi*8 + [0..7]
  v8bf hh = *(const v8bf*)(p + 16);   // K = kbase + 16 + hi*8 + [0..7]
  v16bf r;
#pragma unroll
  for (int i = 0; i < 8; ++i) { r[i] = lo[i]; r[i + 8] = hh[i]; }
  return r;
}

__device__ __forceinline__ v16bf load_b_frag(const bf16_t* base, int rowStride,
                                             int kbase, int lane) {
  const int hi = lane >> 4;
  const bf16_t* p = base + (size_t)(lane & 15) * rowStride + kbase + hi * 8;
  v8bf a = *(const v8bf*)p;           // K = kbase + hi*8 + [0..7]
  v8bf b = *(const v8bf*)(p + 16);    // K = kbase + 16 + hi*8 + [0..7]
  v16bf r;
#pragma unroll
  for (int i = 0; i < 8; ++i) { r[i] = a[i]; r[i + 8] = b[i]; }
  return r;
}

__device__ __forceinline__ v8f wmma_bf16(v16bf a, v16bf b, v8f c) {
  return __builtin_amdgcn_wmma_f32_16x16x32_bf16(false, a, false, b,
                                                 (short)0, c, false, false);
}

#if USE_TDM
// ---------------------------------------------------------------------------
// Tensor Data Mover: async-load a 128-row x 32-elem bf16 tile (row stride
// `lda` elements in global) into LDS. pad_enable inserts 16B after every 64B
// stored, so the LDS tile lands with a 40-element (80B) row stride — exactly
// the layout load_a_frag() expects. D# packing per CDNA5 ISA.
// ---------------------------------------------------------------------------
__device__ __forceinline__ void tdm_load_a_tile(const bf16_t* gptr,
                                                unsigned ldsByteOff, int lda) {
  unsigned long long ga = (unsigned long long)(uintptr_t)gptr;
  v4u_t g0;
  g0[0] = 1u;                                   // count=1 (valid user D#)
  g0[1] = ldsByteOff;                           // lds_addr
  g0[2] = (unsigned)ga;                         // global_addr[31:0]
  g0[3] = (unsigned)((ga >> 32) & 0x1FFFFFFu)   // global_addr[56:32]
        | 0x80000000u;                          // type=2 ("image")
  v8i_t g1;
  g1[0] = (1 << 16)     // data_size = 2 bytes
        | (1 << 20)     // pad_enable
        | (3 << 22)     // pad_interval = 16 DWORDs (64B)
        | (3 << 25);    // pad_amount   = 4 DWORDs (16B)  -> 80B LDS row stride
  g1[1] = (32  << 16);  // tensor_dim0 = 32 (lo16 at bits[63:48])
  g1[2] = (128 << 16);  // tensor_dim0 hi16=0 | tensor_dim1 lo16 = 128
  g1[3] = (32  << 16);  // tensor_dim1 hi16=0 | tile_dim0 = 32
  g1[4] = 128;          // tile_dim1 = 128 | tile_dim2 = 0
  g1[5] = lda;          // tensor_dim0_stride lo32 (element units)
  g1[6] = 0;            // stride hi16 | tensor_dim1_stride lo16
  g1[7] = 0;
  v4i_t z4 = {0, 0, 0, 0};                      // groups 2/3 unused (2D tensor)
  v8i_t z8 = {0, 0, 0, 0, 0, 0, 0, 0};          // extended group (unused)
  __builtin_amdgcn_tensor_load_to_lds(g0, g1, z4, z4, z8, 0);
}
#endif

// ---------------------------------------------------------------------------
// f32 -> bf16 convert (grid-stride)
// ---------------------------------------------------------------------------
__global__ void cvt_bf16_kernel(const float* __restrict__ in,
                                bf16_t* __restrict__ out, long long n) {
  long long i = (long long)blockIdx.x * blockDim.x + threadIdx.x;
  long long s = (long long)gridDim.x * blockDim.x;
  for (; i < n; i += s) out[i] = (bf16_t)in[i];
}

// ---------------------------------------------------------------------------
// Generic tiled WMMA GEMM. Block = 256 threads = 8 waves (4 M x 2 N), block
// tile 128(M) x 64(N), K-step 32; each wave computes a 32x32 sub-tile = 4
// WMMAs per K-step with A/B fragment reuse. The A tile is streamed by the
// Tensor Data Mover (double-buffered: issue DMA for step i+1, wait count<=1
// so step i's buffer is complete, barrier publishes it to all waves).
// A is [M][K] (row stride lda).
// BNK=true : B stored [N][K] (weights layout).  BNK=false: B stored [K][N].
// Per-z offsets: off(z) = (z/zmod)*Q + (z%zmod)*R  (element units).
// ---------------------------------------------------------------------------
template <bool BNK, bool OUTF32>
__global__ void __launch_bounds__(256)
gemm_kernel(const bf16_t* __restrict__ A, long long aQ, long long aR, int lda,
            const bf16_t* __restrict__ B, long long bQ, long long bR, int ldb,
            void* __restrict__ Cv, long long cQ, long long cR, int ldc,
            const float* __restrict__ bias, int biasR, int Kdim, int zmod) {
  const int tid = threadIdx.x, lane = tid & 31, w = tid >> 5;
  const int wm = w & 3, wn = w >> 2;
  const int m0 = blockIdx.x * 128, n0 = blockIdx.y * 64;
  const int z = blockIdx.z, zq = z / zmod, zr = z % zmod;
  const bf16_t* Ab = A + (size_t)(zq * aQ + zr * aR);
  const bf16_t* Bb = B + (size_t)(zq * bQ + zr * bR);

  __shared__ __attribute__((aligned(128))) bf16_t As[2][128 * 40];
  __shared__ __attribute__((aligned(128))) bf16_t Bs[64 * 40];

  v8f acc[2][2] = {};

#if USE_TDM
  if (w == 0) {
    tdm_load_a_tile(Ab + (size_t)m0 * lda,
                    (unsigned)(uintptr_t)&As[0][0], lda);
  }
#endif

  for (int k0 = 0; k0 < Kdim; k0 += 32) {
    const int cur = (k0 >> 5) & 1;
#if USE_TDM
    if (w == 0) {
      if (k0 + 32 < Kdim) {
        tdm_load_a_tile(Ab + (size_t)m0 * lda + (k0 + 32),
                        (unsigned)(uintptr_t)&As[cur ^ 1][0], lda);
        __builtin_amdgcn_s_wait_tensorcnt(1);  // tile `cur` complete
      } else {
        __builtin_amdgcn_s_wait_tensorcnt(0);
      }
    }
#else
    {  // cooperative fallback staging of the A tile (two 64-row passes)
      const int r = tid >> 2, ch = tid & 3;
      *(v8bf*)&As[cur][r * 40 + ch * 8] =
          *(const v8bf*)(Ab + (size_t)(m0 + r) * lda + k0 + ch * 8);
      *(v8bf*)&As[cur][(r + 64) * 40 + ch * 8] =
          *(const v8bf*)(Ab + (size_t)(m0 + 64 + r) * lda + k0 + ch * 8);
    }
#endif
    // Stage B tile (64 cols x 32 K) into [n][k] layout.
    if (BNK) {
      const int r = tid >> 2, ch = tid & 3;  // 64 rows x 4 chunks
      *(v8bf*)&Bs[r * 40 + ch * 8] =
          *(const v8bf*)(Bb + (size_t)(n0 + r) * ldb + k0 + ch * 8);
    } else {  // B is [K][N]: transpose while staging (32 k-rows x 64 n)
      const int kr = tid >> 3, ch = tid & 7;
      v8bf t = *(const v8bf*)(Bb + (size_t)(k0 + kr) * ldb + n0 + ch * 8);
#pragma unroll
      for (int j = 0; j < 8; ++j) Bs[(ch * 8 + j) * 40 + kr] = t[j];
    }
    __syncthreads();

    v16bf a0 = load_a_frag(&As[cur][(wm * 32 +  0) * 40], 40, 0, lane);
    v16bf a1 = load_a_frag(&As[cur][(wm * 32 + 16) * 40], 40, 0, lane);
    v16bf b0 = load_b_frag(&Bs[(wn * 32 +  0) * 40], 40, 0, lane);
    v16bf b1 = load_b_frag(&Bs[(wn * 32 + 16) * 40], 40, 0, lane);
    acc[0][0] = wmma_bf16(a0, b0, acc[0][0]);
    acc[0][1] = wmma_bf16(a0, b1, acc[0][1]);
    acc[1][0] = wmma_bf16(a1, b0, acc[1][0]);
    acc[1][1] = wmma_bf16(a1, b1, acc[1][1]);
    __syncthreads();
  }

  const size_t cbase = (size_t)(zq * cQ + zr * cR);
#pragma unroll
  for (int j = 0; j < 2; ++j) {
    const int col = n0 + wn * 32 + j * 16 + (lane & 15);
    const float bv = bias ? bias[biasR * zr + col] : 0.0f;
#pragma unroll
    for (int i2 = 0; i2 < 2; ++i2) {
#pragma unroll
      for (int e = 0; e < 8; ++e) {
        const int mg = m0 + wm * 32 + i2 * 16 + e + (lane >> 4) * 8;
        const float val = acc[i2][j][e] + bv;
        if (OUTF32)
          ((float*)Cv)[cbase + (size_t)mg * ldc + col] = val;
        else
          ((bf16_t*)Cv)[cbase + (size_t)mg * ldc + col] = (bf16_t)val;
      }
    }
  }
}

// ---------------------------------------------------------------------------
// Fused weighted-logits kernel. One wave per (b, 16n, 16m) tile:
//   lw[n,m,g] = log(max(relu(pe[b,n,m,:]·pos_w[g,:] + pos_b[g]), eps))   (WMMA)
//   weighted[b,n,g,m] = lw + (q[b,n,g,:]·k[b,m,g,:]) / 8                 (WMMA)
// pe is streamed once (the 512 MB roofline term), staged coalesced to LDS
// with on-the-fly bf16 conversion. q/k fragments come straight from global
// (L2-resident, 4 MB each).
// ---------------------------------------------------------------------------
__global__ void __launch_bounds__(32)
weighted_kernel(const float* __restrict__ pe, const bf16_t* __restrict__ posw,
                const float* __restrict__ posb, const bf16_t* __restrict__ qb,
                const bf16_t* __restrict__ kb, float* __restrict__ weighted) {
  const int lane = threadIdx.x & 31;
  const int hi = lane >> 4;
  const int n0 = blockIdx.x * 16;
  const int m0 = blockIdx.y * 16;
  const int b = blockIdx.z;

  __shared__ bf16_t Abf[16 * 72];    // pe tile, [16 m][64 e] bf16 (stride 72)
  __shared__ float Ls[256 * 17];     // lw, [(n*16+m)][g] padded

  // pos_w is [G=16][E=64] = B-in-[N][K] layout; two K=32 fragments.
  const v16bf bw0 = load_b_frag(posw, 64, 0, lane);
  const v16bf bw1 = load_b_frag(posw, 64, 32, lane);
  const float pb = posb[lane & 15];

  for (int r = 0; r < 16; ++r) {
    // Stage pe[b, n0+r, m0..m0+15, 0..63] (contiguous 4 KiB f32) -> bf16 LDS.
    const float* src = pe + ((size_t)(b * NN + n0 + r) * MM + m0) * EE;
#pragma unroll
    for (int i = 0; i < 8; ++i) {
      const int idx = i * 32 + lane;          // 256 float4 chunks
      const int mm = idx >> 4, e4 = idx & 15;
      v4f v = *(const v4f*)(src + (size_t)mm * EE + e4 * 4);
      v4bf o;
      o[0] = (bf16_t)v[0]; o[1] = (bf16_t)v[1];
      o[2] = (bf16_t)v[2]; o[3] = (bf16_t)v[3];
      *(v4bf*)&Abf[mm * 72 + e4 * 4] = o;
    }
    __syncthreads();
    v16bf a0 = load_a_frag(Abf, 72, 0, lane);
    v16bf a1 = load_a_frag(Abf, 72, 32, lane);
    v8f acc = {};
    acc = wmma_bf16(a0, bw0, acc);
    acc = wmma_bf16(a1, bw1, acc);
    const int g = lane & 15;
#pragma unroll
    for (int e = 0; e < 8; ++e) {
      const int mrow = e + hi * 8;
      float v = acc[e] + pb;
      v = fmaxf(v, 0.0f);
      v = __logf(fmaxf(v, 1e-6f));
      Ls[(r * 16 + mrow) * 17 + g] = v;
    }
    __syncthreads();
  }

  // q·k^T per group; combine with lw and write coalesced along m.
  const bf16_t* qbase = qb + (size_t)(b * NN + n0) * 1024;
  const bf16_t* kbase = kb + (size_t)(b * MM + m0) * 1024;
  const int mcol = lane & 15;
  for (int g = 0; g < GG; ++g) {
    v8f acc = {};
#pragma unroll
    for (int kk = 0; kk < 64; kk += 32) {
      v16bf aq = load_a_frag(qbase + g * 64, 1024, kk, lane);  // rows = n
      v16bf bk = load_b_frag(kbase + g * 64, 1024, kk, lane);  // [m][d] = [N][K]
      acc = wmma_bf16(aq, bk, acc);
    }
#pragma unroll
    for (int e = 0; e < 8; ++e) {
      const int nrow = e + hi * 8;
      const float w = acc[e] * 0.125f + Ls[(nrow * 16 + mcol) * 17 + g];
      weighted[((size_t)(b * NN + n0 + nrow) * GG + g) * MM + m0 + mcol] = w;
    }
  }
}

// ---------------------------------------------------------------------------
// Softmax over m (row length 1024). One wave per row, 8 rows per block.
// Writes bf16 p in place over the weighted buffer: row i's bf16 data (2 KiB)
// sits at the start of row i's f32 slot (4 KiB) -> no cross-row aliasing.
// ---------------------------------------------------------------------------
__global__ void __launch_bounds__(256)
softmax_kernel(const float* __restrict__ weighted, bf16_t* __restrict__ p) {
  const int lane = threadIdx.x & 31, wid = threadIdx.x >> 5;
  const size_t row = (size_t)blockIdx.x * 8 + wid;
  const float* src = weighted + row * 1024;
  float v[32];
  float mx = -3.4e38f;
#pragma unroll
  for (int i = 0; i < 32; ++i) { v[i] = src[lane + i * 32]; mx = fmaxf(mx, v[i]); }
#pragma unroll
  for (int off = 16; off > 0; off >>= 1) mx = fmaxf(mx, __shfl_xor(mx, off, 32));
  float s = 0.0f;
#pragma unroll
  for (int i = 0; i < 32; ++i) { v[i] = __expf(v[i] - mx); s += v[i]; }
#pragma unroll
  for (int off = 16; off > 0; off >>= 1) s += __shfl_xor(s, off, 32);
  const float inv = 1.0f / s;
  bf16_t* dst = p + row * 2048;  // stride-2048 bf16 rows over the f32 buffer
#pragma unroll
  for (int i = 0; i < 32; ++i) dst[lane + i * 32] = (bf16_t)(v[i] * inv);
}

// ---------------------------------------------------------------------------
// Launch. Inputs (setup_inputs order): roi_feat, position_embedding, pos_w,
// pos_b, q_w, q_b, k_w, k_b, conv_w, conv_b. Output: (B,N,1024) f32.
// Requires ws_size >= ~211 MB.
// ---------------------------------------------------------------------------
extern "C" void kernel_launch(void* const* d_in, const int* in_sizes, int n_in,
                              void* d_out, int out_size, void* d_ws,
                              size_t ws_size, hipStream_t stream) {
  const float* roi  = (const float*)d_in[0];
  const float* pe   = (const float*)d_in[1];
  const float* posw = (const float*)d_in[2];
  const float* posb = (const float*)d_in[3];
  const float* qw   = (const float*)d_in[4];
  const float* qbia = (const float*)d_in[5];
  const float* kw   = (const float*)d_in[6];
  const float* kbia = (const float*)d_in[7];
  const float* cw   = (const float*)d_in[8];
  const float* cbia = (const float*)d_in[9];
  (void)in_sizes; (void)n_in; (void)out_size; (void)ws_size;

  char* ws = (char*)d_ws;
  bf16_t* roiV  = (bf16_t*)(ws + ROIV_OFF);
  bf16_t* qwB   = (bf16_t*)(ws + QW_OFF);
  bf16_t* kwB   = (bf16_t*)(ws + KW_OFF);
  bf16_t* cwB   = (bf16_t*)(ws + CW_OFF);
  bf16_t* poswB = (bf16_t*)(ws + POSW_OFF);
  bf16_t* qB    = (bf16_t*)(ws + QB16_OFF);
  bf16_t* kB    = (bf16_t*)(ws + KB16_OFF);
  float*  wgt   = (float*)(ws + WGT_OFF);
  bf16_t* pbf   = (bf16_t*)(ws + WGT_OFF);   // in-place, stride-2048 rows
  bf16_t* outT  = (bf16_t*)(ws + OUTT_OFF);

  // 1) bf16 conversions
  cvt_bf16_kernel<<<2048, 256, 0, stream>>>(roi, roiV, (long long)BB * NN * FF);
  cvt_bf16_kernel<<<1024, 256, 0, stream>>>(qw, qwB, 1024LL * 1024);
  cvt_bf16_kernel<<<1024, 256, 0, stream>>>(kw, kwB, 1024LL * 1024);
  cvt_bf16_kernel<<<1024, 256, 0, stream>>>(cw, cwB, 1024LL * 1024);
  cvt_bf16_kernel<<<4, 256, 0, stream>>>(posw, poswB, (long long)GG * EE);

  // 2) q = roi @ q_w^T + q_b ; k = roi @ k_w^T + k_b   (M=2048, N=1024, K=1024)
  gemm_kernel<true, false><<<dim3(16, 16, 1), 256, 0, stream>>>(
      roiV, 0LL, 0LL, 1024, qwB, 0LL, 0LL, 1024, qB, 0LL, 0LL, 1024,
      qbia, 0, 1024, 1);
  gemm_kernel<true, false><<<dim3(16, 16, 1), 256, 0, stream>>>(
      roiV, 0LL, 0LL, 1024, kwB, 0LL, 0LL, 1024, kB, 0LL, 0LL, 1024,
      kbia, 0, 1024, 1);

  // 3) weighted = log(max(relu(pe·pos_w + pos_b), eps)) + q·k^T / 8
  weighted_kernel<<<dim3(NN / 16, MM / 16, BB), 32, 0, stream>>>(
      pe, poswB, posb, qB, kB, wgt);

  // 4) softmax over m -> p (bf16, in place)
  softmax_kernel<<<(BB * NN * GG) / 8, 256, 0, stream>>>(wgt, pbf);

  // 5) out_t[b,g] = p[b,:,g,:] @ roi[b]   (32 batched 1024^3, B is [K][N])
  gemm_kernel<false, false><<<dim3(8, 16, BB * GG), 256, 0, stream>>>(
      pbf, 33554432LL, 2048LL, 32768,           // p: b-stride, g-stride, n-stride
      roiV, 1048576LL, 0LL, 1024,               // V = roi[b]
      outT, 16777216LL, 1024LL, 16384,          // out_t (b,n,g,f)
      nullptr, 0, 1024, GG);

  // 6) out[b,n,g*64+o] = out_t[b,n,g,:] @ cw[g,o,:] + conv_b
  gemm_kernel<true, true><<<dim3(8, 1, BB * GG), 256, 0, stream>>>(
      outT, 16777216LL, 1024LL, 16384,
      cwB, 0LL, 65536LL, 1024,                  // cw[g] is [64][1024] = [N][K]
      d_out, 1048576LL, 64LL, 1024,
      cbia, 64, 1024, GG);
}
